// ProteinGAT_20100446945575
// MI455X (gfx1250) — matrix operand order repaired and checked by reading, not compiled, not run on hardware
//
#include <hip/hip_runtime.h>
#include <hip/hip_bf16.h>

// ---------------------------------------------------------------------------
// ProteinGAT pipeline for gfx1250 (MI455X, wave32, WMMA)
//   L1: GATConv(41->128, H=2, concat)  -> relu
//   L2: GATConv(256->256, H=1)         -> relu
//   global mean pool -> fc1(256->1024)+relu -> fc2(1024->128)
// Dense transforms use v_wmma_f32_16x16x32_f16 (f16 in, f32 acc).
// GEMM: one wave computes a 16x64 C tile (4 accumulators) so the A fragment
// is reused 4x and 4 independent WMMA chains hide each other's latency.
// ---------------------------------------------------------------------------

#define N_NODES 50000
#define N_EDGES 800000
#define N_GRAPHS 256
#define F_IN 41
#define F_IN_PAD 64
#define HC 256          // H*C for both GAT layers (2*128 and 1*256)
#define NEG_SLOPE 0.2f

typedef _Float16 v16h __attribute__((ext_vector_type(16)));
typedef _Float16 v8h  __attribute__((ext_vector_type(8)));
typedef float    v8f  __attribute__((ext_vector_type(8)));

// ---------- float <-> order-preserving uint key (for atomic float max) ------
__device__ __forceinline__ unsigned fkey(float f) {
    unsigned u = __float_as_uint(f);
    return (u & 0x80000000u) ? ~u : (u | 0x80000000u);
}
__device__ __forceinline__ float funkey(unsigned u) {
    u = (u & 0x80000000u) ? (u & 0x7FFFFFFFu) : ~u;
    return __uint_as_float(u);
}

// ---------------------------------------------------------------------------
// Pack weight W[K,N] (f32, row-major) into wave32 WMMA B-fragment layout:
//   Bp[(kt*nTiles + nt)*512 + lane*16 + j]
//   lane 0-15  : n = lane,     K = kt*32 + j (j < 8), kt*32 + 8 + j (j >= 8)
//   lane 16-31 : n = lane-16,  K = kt*32 + 8 + j (j < 8), kt*32 + 16 + j (j >= 8)
// Rows k >= Ksrc are zero padding (used for F_IN 41 -> 64).
// ---------------------------------------------------------------------------
__global__ void pack_b_kernel(const float* __restrict__ W, _Float16* __restrict__ Bp,
                              int Kpad, int Ksrc, int N) {
    int idx = blockIdx.x * blockDim.x + threadIdx.x;
    int total = (Kpad >> 5) * (N >> 4) * 512;
    if (idx >= total) return;
    int j    = idx & 15;
    int lane = (idx >> 4) & 31;
    int tile = idx >> 9;
    int nTiles = N >> 4;
    int nt = tile % nTiles;
    int kt = tile / nTiles;
    int n  = nt * 16 + (lane & 15);
    int k  = kt * 32 + ((lane < 16) ? 0 : 8) + j + ((j < 8) ? 0 : 8);
    float v = (k < Ksrc) ? W[(size_t)k * N + n] : 0.0f;
    Bp[idx] = (_Float16)v;
}

// f32 [M,Ksrc] -> f16 [M,Kdst] with zero padding (Kdst >= Ksrc).
__global__ void cast_pad_kernel(const float* __restrict__ X, _Float16* __restrict__ Xh,
                                int M, int Ksrc, int Kdst) {
    long idx = (long)blockIdx.x * blockDim.x + threadIdx.x;
    long total = (long)M * Kdst;
    if (idx >= total) return;
    int col = (int)(idx % Kdst);
    long row = idx / Kdst;
    Xh[idx] = (col < Ksrc) ? (_Float16)X[row * Ksrc + col] : (_Float16)0.0f;
}

// ---------------------------------------------------------------------------
// GEMM: C[M,N] = A[M,K]@B[K,N] (+bias, +relu).  A: f16 row-major.
// Bp: pre-packed fragments. One wave per 16x64 C tile (4 N-tiles).
// EXEC stays all-ones within every live wave (tile id is wave-uniform)
// as WMMA requires. Requires N % 64 == 0, K % 32 == 0, M % 16 == 0.
// ---------------------------------------------------------------------------
__global__ void gemm_wmma_kernel(const _Float16* __restrict__ A,
                                 const _Float16* __restrict__ Bp,
                                 const float* __restrict__ bias,
                                 float* __restrict__ C,
                                 int M, int N, int K, int relu) {
    int gwave = (int)((blockIdx.x * blockDim.x + threadIdx.x) >> 5);
    int lane  = threadIdx.x & 31;
    int nTiles  = N >> 4;
    int nGroups = nTiles >> 2;              // 4 N-tiles per wave
    int mTiles  = M >> 4;
    if (gwave >= mTiles * nGroups) return;
    int mt  = gwave / nGroups;
    int ng  = gwave % nGroups;
    int nt0 = ng << 2;

    int mrowA = mt * 16 + (lane & 15);      // A-fragment row for this lane
    int kb    = (lane < 16) ? 0 : 8;        // A-fragment K phase (ISA 7.12.2)
    const _Float16* Arow = A + (size_t)mrowA * K + kb;
    const _Float16* Bt   = Bp + ((size_t)nt0 * 512) + (size_t)lane * 16;
    size_t bStride = (size_t)nTiles * 512;

    v8f acc0 = {}, acc1 = {}, acc2 = {}, acc3 = {};
    int kTiles = K >> 5;
    for (int kt = 0; kt < kTiles; ++kt) {
        v8h alo = *(const v8h*)(Arow + kt * 32);        // K = kb..kb+7
        v8h ahi = *(const v8h*)(Arow + kt * 32 + 16);   // K = kb+16..kb+23
        v16h a;
#pragma unroll
        for (int i = 0; i < 8; ++i) { a[i] = alo[i]; a[8 + i] = ahi[i]; }
        const _Float16* bk = Bt + (size_t)kt * bStride;
        v16h b0 = *(const v16h*)(bk);
        v16h b1 = *(const v16h*)(bk + 512);
        v16h b2 = *(const v16h*)(bk + 1024);
        v16h b3 = *(const v16h*)(bk + 1536);
        acc0 = __builtin_amdgcn_wmma_f32_16x16x32_f16(false, a, false, b0, (short)0, acc0, false, false);
        acc1 = __builtin_amdgcn_wmma_f32_16x16x32_f16(false, a, false, b1, (short)0, acc1, false, false);
        acc2 = __builtin_amdgcn_wmma_f32_16x16x32_f16(false, a, false, b2, (short)0, acc2, false, false);
        acc3 = __builtin_amdgcn_wmma_f32_16x16x32_f16(false, a, false, b3, (short)0, acc3, false, false);
    }

    int nb   = lane & 15;
    int mrow = mt * 16 + ((lane >= 16) ? 8 : 0);
    v8f accs[4] = { acc0, acc1, acc2, acc3 };
#pragma unroll
    for (int u = 0; u < 4; ++u) {
        int n = (nt0 + u) * 16 + nb;
        float bv = bias ? bias[n] : 0.0f;
#pragma unroll
        for (int r = 0; r < 8; ++r) {
            float v = accs[u][r] + bv;
            if (relu) v = v > 0.0f ? v : 0.0f;
            C[(size_t)(mrow + r) * N + n] = v;
        }
    }
}

// ---------------------------------------------------------------------------
// Per-node attention logits: al_src[n,h] = sum_c h[n,h*C+c]*a_src[h,c]  (H*C==256)
// ---------------------------------------------------------------------------
__global__ void node_logits_kernel(const float* __restrict__ Hf,
                                   const float* __restrict__ a_src,
                                   const float* __restrict__ a_dst,
                                   float* __restrict__ al_src,
                                   float* __restrict__ al_dst,
                                   int Hh, int C) {
    int n = blockIdx.x;
    int t = threadIdx.x;
    __shared__ float ss[HC];
    __shared__ float sd[HC];
    float hv = Hf[(size_t)n * HC + t];
    ss[t] = hv * a_src[t];
    sd[t] = hv * a_dst[t];
    __syncthreads();
    for (int off = C >> 1; off > 0; off >>= 1) {
        if ((t & (C - 1)) < off) { ss[t] += ss[t + off]; sd[t] += sd[t + off]; }
        __syncthreads();
    }
    if ((t & (C - 1)) == 0) {
        int h = t / C;
        al_src[n * Hh + h] = ss[t];
        al_dst[n * Hh + h] = sd[t];
    }
}

// Pass 1: leaky-relu logits + per-(dst,head) segment max via keyed atomicMax.
__global__ void edge_max_kernel(const int* __restrict__ src, const int* __restrict__ dst,
                                const float* __restrict__ al_src, const float* __restrict__ al_dst,
                                float* __restrict__ elog, unsigned* __restrict__ mmax, int Hh) {
    int e = blockIdx.x * blockDim.x + threadIdx.x;
    if (e >= N_EDGES + N_NODES) return;
    int s = (e < N_EDGES) ? src[e] : (e - N_EDGES);   // appended self-loops
    int d = (e < N_EDGES) ? dst[e] : (e - N_EDGES);
    for (int h = 0; h < Hh; ++h) {
        float l = al_src[s * Hh + h] + al_dst[d * Hh + h];
        l = (l > 0.0f) ? l : NEG_SLOPE * l;
        elog[(size_t)e * Hh + h] = l;
        atomicMax(&mmax[d * Hh + h], fkey(l));
    }
}

// Pass 2: exp(l - max) in place + per-(dst,head) denom.
__global__ void edge_exp_kernel(const int* __restrict__ src, const int* __restrict__ dst,
                                float* __restrict__ elog, const unsigned* __restrict__ mmax,
                                float* __restrict__ denom, int Hh) {
    int e = blockIdx.x * blockDim.x + threadIdx.x;
    if (e >= N_EDGES + N_NODES) return;
    int d = (e < N_EDGES) ? dst[e] : (e - N_EDGES);
    for (int h = 0; h < Hh; ++h) {
        float m  = funkey(mmax[d * Hh + h]);
        float ex = __expf(elog[(size_t)e * Hh + h] - m);
        elog[(size_t)e * Hh + h] = ex;
        atomicAdd(&denom[d * Hh + h], ex);
    }
}

// Pass 3: out[dst, :] += alpha * h[src, :]. Block per edge, 256 ch = H*C.
__global__ void edge_scatter_kernel(const int* __restrict__ src, const int* __restrict__ dst,
                                    const float* __restrict__ Hf,
                                    const float* __restrict__ elog,
                                    const float* __restrict__ denom,
                                    float* __restrict__ out, int Hh, int C) {
    int e = blockIdx.x;
    int t = threadIdx.x;
    int s = (e < N_EDGES) ? src[e] : (e - N_EDGES);
    int d = (e < N_EDGES) ? dst[e] : (e - N_EDGES);
    int h = t / C;
    float w = elog[(size_t)e * Hh + h] / (denom[d * Hh + h] + 1e-16f);
    atomicAdd(&out[(size_t)d * HC + t], w * Hf[(size_t)s * HC + t]);
}

// relu(out + bias) -> f16 (feeds the next WMMA GEMM).
__global__ void bias_relu_cast_kernel(const float* __restrict__ out,
                                      const float* __restrict__ bias,
                                      _Float16* __restrict__ Xh, long total) {
    long idx = (long)blockIdx.x * blockDim.x + threadIdx.x;
    if (idx >= total) return;
    float v = out[idx] + bias[idx & (HC - 1)];
    Xh[idx] = (_Float16)(v > 0.0f ? v : 0.0f);
}

// relu(out2 + b2), then segment-sum into graphs (+ node counts).
__global__ void pool_kernel(const float* __restrict__ out2, const float* __restrict__ b2,
                            const int* __restrict__ batch,
                            float* __restrict__ sums, float* __restrict__ cnts) {
    long idx = (long)blockIdx.x * blockDim.x + threadIdx.x;
    if (idx >= (long)N_NODES * HC) return;
    int n = (int)(idx >> 8);
    int c = (int)(idx & (HC - 1));
    float v = out2[idx] + b2[c];
    v = v > 0.0f ? v : 0.0f;
    int g = batch[n];
    atomicAdd(&sums[(size_t)g * HC + c], v);
    if (c == 0) atomicAdd(&cnts[g], 1.0f);
}

// g = sums / max(cnt,1) -> f16 for the MLP GEMM.
__global__ void pool_finalize_kernel(const float* __restrict__ sums,
                                     const float* __restrict__ cnts,
                                     _Float16* __restrict__ Gh) {
    int idx = blockIdx.x * blockDim.x + threadIdx.x;
    if (idx >= N_GRAPHS * HC) return;
    float cnt = cnts[idx >> 8];
    Gh[idx] = (_Float16)(sums[idx] / fmaxf(cnt, 1.0f));
}

// ---------------------------------------------------------------------------
extern "C" void kernel_launch(void* const* d_in, const int* in_sizes, int n_in,
                              void* d_out, int out_size, void* d_ws, size_t ws_size,
                              hipStream_t stream) {
    const float* x      = (const float*)d_in[0];
    const int*   src    = (const int*)d_in[1];
    const int*   dst    = (const int*)d_in[2];
    const int*   batch  = (const int*)d_in[3];
    const float* W1     = (const float*)d_in[4];
    const float* a_src1 = (const float*)d_in[5];
    const float* a_dst1 = (const float*)d_in[6];
    const float* b1     = (const float*)d_in[7];
    const float* W2     = (const float*)d_in[8];
    const float* a_src2 = (const float*)d_in[9];
    const float* a_dst2 = (const float*)d_in[10];
    const float* b2     = (const float*)d_in[11];
    const float* fc1_w  = (const float*)d_in[12];
    const float* fc1_b  = (const float*)d_in[13];
    const float* fc2_w  = (const float*)d_in[14];
    const float* fc2_b  = (const float*)d_in[15];
    float* out = (float*)d_out;

    // ---- carve workspace (deterministic offsets, 256B aligned) ----
    char* base = (char*)d_ws;
    size_t off = 0;
    auto carve = [&](size_t bytes) -> char* {
        char* p = base + off;
        off = (off + bytes + 255) & ~(size_t)255;
        return p;
    };
    float*      h_feat  = (float*)carve((size_t)N_NODES * HC * 4);     // GEMM output h
    float*      out_agg = (float*)carve((size_t)N_NODES * HC * 4);     // scatter target
    _Float16*   xh      = (_Float16*)carve((size_t)N_NODES * HC * 2);  // f16 GEMM input
    float*      elog    = (float*)carve((size_t)(N_EDGES + N_NODES) * 2 * 4);
    unsigned*   mmax    = (unsigned*)carve((size_t)N_NODES * 2 * 4);
    float*      denom   = (float*)carve((size_t)N_NODES * 2 * 4);
    float*      alsrc   = (float*)carve((size_t)N_NODES * 2 * 4);
    float*      aldst   = (float*)carve((size_t)N_NODES * 2 * 4);
    _Float16*   W1p     = (_Float16*)carve((size_t)F_IN_PAD * 256 * 2);
    _Float16*   W2p     = (_Float16*)carve((size_t)256 * 256 * 2);
    _Float16*   fc1p    = (_Float16*)carve((size_t)256 * 1024 * 2);
    _Float16*   fc2p    = (_Float16*)carve((size_t)1024 * 128 * 2);
    float*      sums    = (float*)carve((size_t)N_GRAPHS * HC * 4);
    float*      cnts    = (float*)carve((size_t)N_GRAPHS * 4);
    _Float16*   gh      = (_Float16*)carve((size_t)N_GRAPHS * HC * 2);
    float*      q1      = (float*)carve((size_t)N_GRAPHS * 1024 * 4);
    _Float16*   q1h     = (_Float16*)carve((size_t)N_GRAPHS * 1024 * 2);

    const int E2 = N_EDGES + N_NODES;
    auto cdiv = [](long a, long b) { return (int)((a + b - 1) / b); };

    // ---- pack weights into WMMA B-fragment layout ----
    pack_b_kernel<<<cdiv((F_IN_PAD/32)*(256/16)*512, 256), 256, 0, stream>>>(W1, W1p, F_IN_PAD, F_IN, 256);
    pack_b_kernel<<<cdiv((256/32)*(256/16)*512, 256), 256, 0, stream>>>(W2, W2p, 256, 256, 256);
    pack_b_kernel<<<cdiv((256/32)*(1024/16)*512, 256), 256, 0, stream>>>(fc1_w, fc1p, 256, 256, 1024);
    pack_b_kernel<<<cdiv((1024/32)*(128/16)*512, 256), 256, 0, stream>>>(fc2_w, fc2p, 1024, 1024, 128);

    // ---- Layer 1: h = pad(x) @ W1 via WMMA ----
    cast_pad_kernel<<<cdiv((long)N_NODES * F_IN_PAD, 256), 256, 0, stream>>>(x, xh, N_NODES, F_IN, F_IN_PAD);
    {
        long waves = ((long)N_NODES / 16) * (256 / 64);   // 16x64 tile per wave
        gemm_wmma_kernel<<<cdiv(waves * 32, 256), 256, 0, stream>>>(xh, W1p, nullptr, h_feat,
                                                                    N_NODES, 256, F_IN_PAD, 0);
    }
    node_logits_kernel<<<N_NODES, HC, 0, stream>>>(h_feat, a_src1, a_dst1, alsrc, aldst, 2, 128);
    hipMemsetAsync(mmax, 0, (size_t)N_NODES * 2 * 4, stream);
    hipMemsetAsync(denom, 0, (size_t)N_NODES * 2 * 4, stream);
    edge_max_kernel<<<cdiv(E2, 256), 256, 0, stream>>>(src, dst, alsrc, aldst, elog, mmax, 2);
    edge_exp_kernel<<<cdiv(E2, 256), 256, 0, stream>>>(src, dst, elog, mmax, denom, 2);
    hipMemsetAsync(out_agg, 0, (size_t)N_NODES * HC * 4, stream);
    edge_scatter_kernel<<<E2, HC, 0, stream>>>(src, dst, h_feat, elog, denom, out_agg, 2, 128);
    bias_relu_cast_kernel<<<cdiv((long)N_NODES * HC, 256), 256, 0, stream>>>(out_agg, b1, xh, (long)N_NODES * HC);

    // ---- Layer 2: h = x2 @ W2 via WMMA ----
    {
        long waves = ((long)N_NODES / 16) * (256 / 64);
        gemm_wmma_kernel<<<cdiv(waves * 32, 256), 256, 0, stream>>>(xh, W2p, nullptr, h_feat,
                                                                    N_NODES, 256, 256, 0);
    }
    node_logits_kernel<<<N_NODES, HC, 0, stream>>>(h_feat, a_src2, a_dst2, alsrc, aldst, 1, 256);
    hipMemsetAsync(mmax, 0, (size_t)N_NODES * 4, stream);
    hipMemsetAsync(denom, 0, (size_t)N_NODES * 4, stream);
    edge_max_kernel<<<cdiv(E2, 256), 256, 0, stream>>>(src, dst, alsrc, aldst, elog, mmax, 1);
    edge_exp_kernel<<<cdiv(E2, 256), 256, 0, stream>>>(src, dst, elog, mmax, denom, 1);
    hipMemsetAsync(out_agg, 0, (size_t)N_NODES * HC * 4, stream);
    edge_scatter_kernel<<<E2, HC, 0, stream>>>(src, dst, h_feat, elog, denom, out_agg, 1, 256);

    // ---- global mean pool ----
    hipMemsetAsync(sums, 0, (size_t)N_GRAPHS * HC * 4, stream);
    hipMemsetAsync(cnts, 0, (size_t)N_GRAPHS * 4, stream);
    pool_kernel<<<cdiv((long)N_NODES * HC, 256), 256, 0, stream>>>(out_agg, b2, batch, sums, cnts);
    pool_finalize_kernel<<<cdiv(N_GRAPHS * HC, 256), 256, 0, stream>>>(sums, cnts, gh);

    // ---- MLP head: fc1 (+relu) then fc2, both WMMA ----
    {
        long waves = (N_GRAPHS / 16) * (1024 / 64);
        gemm_wmma_kernel<<<cdiv(waves * 32, 256), 256, 0, stream>>>(gh, fc1p, fc1_b, q1,
                                                                    N_GRAPHS, 1024, 256, 1);
    }
    cast_pad_kernel<<<cdiv((long)N_GRAPHS * 1024, 256), 256, 0, stream>>>(q1, q1h, N_GRAPHS, 1024, 1024);
    {
        long waves = (N_GRAPHS / 16) * (128 / 64);
        gemm_wmma_kernel<<<cdiv(waves * 32, 256), 256, 0, stream>>>(q1h, fc2p, fc2_b, out,
                                                                    N_GRAPHS, 128, 1024, 0);
    }
    (void)in_sizes; (void)n_in; (void)out_size; (void)ws_size;
}
